// model_87771951661057
// MI455X (gfx1250) — compile-verified, matrix operand order checked by reading, not run on hardware
//
#include <hip/hip_runtime.h>
#include <hip/hip_bf16.h>

#define NB 8
#define NC 64
#define NH 128
#define NW 128
#define NE 8
#define NHID 128
#define HWSZ (NH*NW)            /* 16384 */
#define NTOK (NB*NH*NW)         /* 131072 */
#define NPAIR 56                /* ordered (i0,i1), i0 != i1 -> p = i0*8+i1 */
#define MAXTILES (NTOK/16 + NPAIR)   /* 8248 */
#define EPSF 2.220446049250313e-16f

typedef __attribute__((ext_vector_type(16))) _Float16 v16h;
typedef __attribute__((ext_vector_type(8)))  float    v8f;

union F16Frag { v16h v; uint4 q[2]; };

// Branch-free exact-GELU: erf via Abramowitz-Stegun 7.1.26 (|err|<=1.5e-7).
// Result is cast to f16 before the next WMMA, so approx error is invisible.
__device__ __forceinline__ float gelu_f(float v) {
    float x  = v * 0.70710678118654752f;
    float ax = __builtin_fabsf(x);
    float t  = __builtin_amdgcn_rcpf(__builtin_fmaf(0.3275911f, ax, 1.f));
    float p  = t * __builtin_fmaf(t,
                   __builtin_fmaf(t,
                     __builtin_fmaf(t,
                       __builtin_fmaf(t, 1.061405429f, -1.453152027f),
                       1.421413741f),
                     -0.284496736f),
                   0.254829592f);
    float er = 1.f - p * __expf(-ax * ax);
    er = __builtin_copysignf(er, x);
    float h = 0.5f * v;
    return __builtin_fmaf(h, er, h);
}

// ---------------------------------------------------------------------------
// K0: weight transpose/convert to f16, prompt logits, zero pair counters
//   fc1t[e][n][c] = fc1_w[e][c][n]   (E, HID, C)  -> contiguous-K rows
//   fc2t[e][n][k] = fc2_w[e][k][n]   (E, C, HID)
// ---------------------------------------------------------------------------
__global__ void prep_kernel(const float* __restrict__ fc1_w,
                            const float* __restrict__ fc2_w,
                            const float* __restrict__ prompt,
                            const float* __restrict__ w_g,
                            _Float16* __restrict__ fc1t,
                            _Float16* __restrict__ fc2t,
                            float* __restrict__ plog,
                            int* __restrict__ pcnt) {
    int tid  = blockIdx.x * blockDim.x + threadIdx.x;
    int nthr = gridDim.x * blockDim.x;
    for (int i = tid; i < NE * NHID * NC; i += nthr) {
        int c = i % NC;
        int n = (i / NC) % NHID;
        int e = i / (NC * NHID);
        fc1t[i] = (_Float16)fc1_w[(e * NC + c) * NHID + n];
    }
    for (int i = tid; i < NE * NC * NHID; i += nthr) {
        int k = i % NHID;
        int n = (i / NHID) % NC;
        int e = i / (NC * NHID);
        fc2t[i] = (_Float16)fc2_w[(e * NHID + k) * NC + n];
    }
    if (tid < NB * NE) {
        int e = tid % NE;
        int b = tid / NE;
        float s = 0.f;
        for (int c = 0; c < NC; ++c)
            s += prompt[b * NC + c] * w_g[(NC + c) * NE + e];
        plog[tid] = s;
    }
    if (tid < 64) pcnt[tid] = 0;
}

// ---------------------------------------------------------------------------
// K1: gating. One thread per token. Coalesced strided x reads (lane = w).
// Writes token-major f16 activations xh[t][c], gate records, per-pair counts,
// and per-block loss partials (deterministic reduction later).
// ---------------------------------------------------------------------------
__global__ void gate_kernel(const float* __restrict__ x,
                            const float* __restrict__ w_g,
                            const float* __restrict__ plog,
                            _Float16* __restrict__ xh,
                            int4* __restrict__ gates,
                            int* __restrict__ pcnt,
                            float* __restrict__ partials) {
    __shared__ float wgs[NC * NE];
    __shared__ float swsum[NE];
    __shared__ float sload[NE];
    __shared__ int   spc[NPAIR + 8];   // indexed by p = i0*8+i1 (<64)
    int tid = threadIdx.x;
    for (int i = tid; i < NC * NE; i += blockDim.x) wgs[i] = w_g[i];
    if (tid < NE) { swsum[tid] = 0.f; sload[tid] = 0.f; }
    if (tid < 64) spc[tid] = 0;
    __syncthreads();

    int t  = blockIdx.x * blockDim.x + tid;
    int b  = t >> 14;
    int hw = t & (HWSZ - 1);

    float lg[NE];
#pragma unroll
    for (int e = 0; e < NE; ++e) lg[e] = plog[b * NE + e];

    union { _Float16 h[NC]; uint4 q[8]; } xl;
    const float* xb = x + (size_t)b * NC * HWSZ + hw;
#pragma unroll
    for (int c = 0; c < NC; ++c) {
        float v = xb[(size_t)c * HWSZ];
        xl.h[c] = (_Float16)v;
#pragma unroll
        for (int e = 0; e < NE; ++e) lg[e] += v * wgs[c * NE + e];
    }
    uint4* dst = reinterpret_cast<uint4*>(xh + (size_t)t * NC);
#pragma unroll
    for (int i = 0; i < 8; ++i) dst[i] = xl.q[i];

    // top-2 (ties -> lower index, matches jax top_k)
    float v0 = -3.4e38f, v1 = -3.4e38f;
    int   i0 = 0, i1 = 0;
#pragma unroll
    for (int e = 0; e < NE; ++e) {
        float l = lg[e];
        if (l > v0)      { v1 = v0; i1 = i0; v0 = l; i0 = e; }
        else if (l > v1) { v1 = l; i1 = e; }
    }
    float e1 = expf(v1 - v0);
    float inv = 1.f / (1.f + e1);
    float g0 = inv, g1 = e1 * inv;
    gates[t] = make_int4(i0, i1, __float_as_int(g0), __float_as_int(g1));

    atomicAdd(&spc[i0 * 8 + i1], 1);
    atomicAdd(&swsum[i0], g0);
    atomicAdd(&swsum[i1], g1);
    atomicAdd(&sload[i0], 1.f);
    if (g1 > 0.f) atomicAdd(&sload[i1], 1.f);
    __syncthreads();
    if (tid < NE) {
        partials[blockIdx.x * 16 + tid]      = swsum[tid];
        partials[blockIdx.x * 16 + 8 + tid]  = sload[tid];
    }
    if (tid < 64 && spc[tid] > 0) atomicAdd(&pcnt[tid], spc[tid]);
}

// ---------------------------------------------------------------------------
// K2: scan pair counts -> element offsets + tile prefix; zero cursors
// ---------------------------------------------------------------------------
__global__ void scan_kernel(const int* __restrict__ pcnt,
                            int* __restrict__ offs,
                            int* __restrict__ tileP,
                            int* __restrict__ cursor) {
    if (threadIdx.x == 0 && blockIdx.x == 0) {
        int run = 0, trun = 0;
        for (int i0 = 0; i0 < NE; ++i0)
            for (int i1 = 0; i1 < NE; ++i1) {
                if (i0 == i1) continue;
                int p = i0 * 8 + i1;
                offs[p]  = run;
                tileP[p] = trun;
                run  += pcnt[p];
                trun += (pcnt[p] + 15) >> 4;
            }
        // mark diagonal slots invalid so the linear owner-scan skips them
        for (int d = 0; d < NE; ++d) { offs[d * 9] = -1; }
        tileP[64] = trun;
    }
    if (blockIdx.x == 0 && threadIdx.x < 64) cursor[threadIdx.x] = 0;
}

// ---------------------------------------------------------------------------
// K3: place tokens into their pair list (slot order nondeterministic, but
// every token's result is independent -> output deterministic)
// ---------------------------------------------------------------------------
__global__ void place_kernel(const int4* __restrict__ gates,
                             const int* __restrict__ offs,
                             int* __restrict__ cursor,
                             int* __restrict__ ent) {
    int t = blockIdx.x * blockDim.x + threadIdx.x;
    int4 gr = gates[t];
    int p = gr.x * 8 + gr.y;
    int pos = atomicAdd(&cursor[p], 1);
    ent[offs[p] + pos] = t;
}

// ---------------------------------------------------------------------------
// K4: deterministic loss reduction
// ---------------------------------------------------------------------------
__global__ void loss_kernel(const float* __restrict__ partials,
                            float* __restrict__ out_loss) {
    __shared__ float red[16];
    int tid = threadIdx.x;
    if (tid < 16) {
        float s = 0.f;
        for (int blk = 0; blk < NTOK / 256; ++blk) s += partials[blk * 16 + tid];
        red[tid] = s;
    }
    __syncthreads();
    if (tid == 0) {
        float mw = 0.f, ml = 0.f;
        for (int e = 0; e < NE; ++e) { mw += red[e]; ml += red[8 + e]; }
        mw *= (1.f / NE); ml *= (1.f / NE);
        float vw = 0.f, vl = 0.f;
        for (int e = 0; e < NE; ++e) {
            float d = red[e] - mw;       vw += d * d;
            float d2 = red[8 + e] - ml;  vl += d2 * d2;
        }
        vw *= (1.f / (NE - 1)); vl *= (1.f / (NE - 1));
        *out_loss = vw / (mw * mw + 1e-10f) + vl / (ml * ml + 1e-10f);
    }
}

// ---------------------------------------------------------------------------
// K5: sparse expert compute. One wave = one 16-token tile of one expert PAIR.
// Runs both experts of the pair (64 static WMMAs), ensembles, logs, writes
// token-major yt with 64B-coalesced stores.
// ---------------------------------------------------------------------------
__global__ __launch_bounds__(256) void expert_kernel(
        const _Float16* __restrict__ xh,
        const int4* __restrict__ gates,
        const int* __restrict__ ent,
        const int* __restrict__ pcnt,
        const int* __restrict__ offs,
        const int* __restrict__ tileP,
        const _Float16* __restrict__ fc1t,   // (E, HID, C) f16
        const _Float16* __restrict__ fc2t,   // (E, C, HID) f16
        const float* __restrict__ fc1_b,
        const float* __restrict__ fc2_b,
        float* __restrict__ yt) {
    __shared__ __align__(16) _Float16 hidS[8][16][NHID]; // 32 KB per-wave tiles
    __shared__ int   tTok[8][16];
    __shared__ float tGa[8][16];
    __shared__ float tGb[8][16];

    int tid  = threadIdx.x;
    int wave = tid >> 5;
    int lane = tid & 31;
    int lr   = lane & 15;
    int lh   = lane >> 4;
    int tileId = blockIdx.x * 8 + wave;
    int total  = tileP[64];
    if (tileId >= total) return;   // wave-granular exit, no block barriers used

    // owner scan: tileP is monotone over valid (non-diagonal) slots in p-order;
    // a non-empty owner strictly increases the prefix, so "last q with
    // tileP[q] <= tileId" is always the owning pair.
    int p = 0;
    for (int q = 1; q < 64; ++q) {
        if (offs[q] >= 0 && tileP[q] <= tileId) p = q;
    }
    int e0 = p >> 3, e1 = p & 7;
    int lt  = tileId - tileP[p];
    int cnt = pcnt[p];
    int base = offs[p] + lt * 16;

    if (lh == 0) {
        int li = lt * 16 + lr;
        int tok = -1; float ga = 0.f, gb = 0.f;
        if (li < cnt) {
            tok = ent[base + lr];
            int4 gr = gates[tok];
            ga = __int_as_float(gr.z);   // gate for i0 == e0
            gb = __int_as_float(gr.w);   // gate for i1 == e1
        }
        tTok[wave][lr] = tok;
        tGa[wave][lr]  = ga;
        tGb[wave][lr]  = gb;
    }
    // same-wave LDS is in-order; cross-lane read is safe in lockstep
    int tokA = tTok[wave][lr];
    if (tokA < 0) tokA = 0;

    // A fragments (16x32 f16): lane row M=lr; halves K=lh*8..+7 then K+16
    F16Frag A0, A1;
    {
        const _Float16* ab = xh + (size_t)tokA * NC + lh * 8;
        A0.q[0] = *(const uint4*)(ab);
        A0.q[1] = *(const uint4*)(ab + 16);
        A1.q[0] = *(const uint4*)(ab + 32);
        A1.q[1] = *(const uint4*)(ab + 48);
    }

    v8f zero = {};
    v8f ens[4];
#pragma unroll
    for (int i = 0; i < 4; ++i) ens[i] = zero;

#pragma unroll
    for (int s = 0; s < 2; ++s) {
        int e = s ? e1 : e0;
        // ---- MLP1: hid(16x128) = A @ fc1[e], bias + exact GELU ----
        const _Float16* wb = fc1t + (size_t)e * NHID * NC;
#pragma unroll
        for (int nt = 0; nt < 8; ++nt) {
            int n = nt * 16 + lr;
            const _Float16* bb = wb + (size_t)n * NC + lh * 8;
            F16Frag B0, B1;
            B0.q[0] = *(const uint4*)(bb);
            B0.q[1] = *(const uint4*)(bb + 16);
            B1.q[0] = *(const uint4*)(bb + 32);
            B1.q[1] = *(const uint4*)(bb + 48);
            v8f acc = zero;
            acc = __builtin_amdgcn_wmma_f32_16x16x32_f16(
                      false, A0.v, false, B0.v, (short)0, acc, false, false);
            acc = __builtin_amdgcn_wmma_f32_16x16x32_f16(
                      false, A1.v, false, B1.v, (short)0, acc, false, false);
            float bias = fc1_b[e * NHID + n];
#pragma unroll
            for (int j = 0; j < 8; ++j) {
                float v = gelu_f(acc[j] + bias);
                hidS[wave][j + lh * 8][n] = (_Float16)v;
            }
        }
        // re-load hid as A fragments (4 K-blocks of 32)
        F16Frag H[4];
#pragma unroll
        for (int kb = 0; kb < 4; ++kb) {
            const _Float16* hb = &hidS[wave][lr][kb * 32 + lh * 8];
            H[kb].q[0] = *(const uint4*)(hb);
            H[kb].q[1] = *(const uint4*)(hb + 16);
        }
        // ---- MLP2: out(16x64) = hid @ fc2[e], ensemble accumulate ----
        const _Float16* w2 = fc2t + (size_t)e * NC * NHID;
#pragma unroll
        for (int nt2 = 0; nt2 < 4; ++nt2) {
            int n2 = nt2 * 16 + lr;
            const _Float16* rb = w2 + (size_t)n2 * NHID + lh * 8;
            v8f acc = zero;
#pragma unroll
            for (int kb = 0; kb < 4; ++kb) {
                F16Frag Bf;
                Bf.q[0] = *(const uint4*)(rb + kb * 32);
                Bf.q[1] = *(const uint4*)(rb + kb * 32 + 16);
                acc = __builtin_amdgcn_wmma_f32_16x16x32_f16(
                          false, H[kb].v, false, Bf.v, (short)0, acc, false, false);
            }
            float bias = fc2_b[e * NC + n2];
#pragma unroll
            for (int j = 0; j < 8; ++j) {
                int row = j + lh * 8;
                float g = s ? tGb[wave][row] : tGa[wave][row];
                float contrib = (g > 0.f) ? g * __expf(acc[j] + bias) : 0.f;
                ens[nt2][j] += contrib;
            }
        }
    }

    // ---- yt[tok][c] = log(ens or EPS); 64B-coalesced per row ----
#pragma unroll
    for (int nt2 = 0; nt2 < 4; ++nt2) {
        int c = nt2 * 16 + lr;
#pragma unroll
        for (int j = 0; j < 8; ++j) {
            int row = j + lh * 8;
            int tok = tTok[wave][row];
            if (tok >= 0) {
                float v = ens[nt2][j];
                v = (v == 0.f) ? EPSF : v;
                yt[(size_t)tok * NC + c] = __logf(v);
            }
        }
    }
}

// ---------------------------------------------------------------------------
// K6: token-major yt -> NCHW y via LDS 64x64 tile transpose (fully coalesced)
// ---------------------------------------------------------------------------
__global__ __launch_bounds__(256) void transpose_kernel(
        const float* __restrict__ yt, float* __restrict__ y) {
    __shared__ float ts[64][65];
    int tid = threadIdx.x;
    int t0  = blockIdx.x * 64;
    int b   = t0 >> 14;
    int hw0 = t0 & (HWSZ - 1);
#pragma unroll
    for (int i = 0; i < 16; ++i) {
        int idx = tid + i * 256;
        int r = idx >> 6, c = idx & 63;
        ts[r][c] = yt[(size_t)(t0 + r) * NC + c];
    }
    __syncthreads();
#pragma unroll
    for (int i = 0; i < 16; ++i) {
        int idx = tid + i * 256;
        int c = idx >> 6, r = idx & 63;
        y[((size_t)(b * NC + c) << 14) + hw0 + r] = ts[r][c];
    }
}

// ---------------------------------------------------------------------------
extern "C" void kernel_launch(void* const* d_in, const int* in_sizes, int n_in,
                              void* d_out, int out_size, void* d_ws, size_t ws_size,
                              hipStream_t stream) {
    const float* x      = (const float*)d_in[0];
    const float* prompt = (const float*)d_in[1];
    const float* w_g    = (const float*)d_in[2];
    const float* fc1_w  = (const float*)d_in[3];
    const float* fc1_b  = (const float*)d_in[4];
    const float* fc2_w  = (const float*)d_in[5];
    const float* fc2_b  = (const float*)d_in[6];

    float* y    = (float*)d_out;
    float* loss = y + (size_t)NTOK * NC;

    // workspace layout (~53 MB)
    char* ws = (char*)d_ws;
    size_t o = 0;
    _Float16* xh    = (_Float16*)(ws + o); o += (size_t)NTOK * NC * 2;        // 16 MB
    int4*     gates = (int4*)    (ws + o); o += (size_t)NTOK * 16;            //  2 MB
    _Float16* fc1t  = (_Float16*)(ws + o); o += (size_t)NE * NHID * NC * 2;
    _Float16* fc2t  = (_Float16*)(ws + o); o += (size_t)NE * NC * NHID * 2;
    float*    plog  = (float*)   (ws + o); o += 256;
    float*    yt    = (float*)   (ws + o); o += (size_t)NTOK * NC * 4;        // 33.5 MB
    int*      ent   = (int*)     (ws + o); o += (size_t)(NTOK + NPAIR * 16) * 4;
    int*      pcnt  = (int*)     (ws + o); o += 256;
    int*      offs  = (int*)     (ws + o); o += 256;
    int*      tileP = (int*)     (ws + o); o += 512;   // 65 ints
    int*      cursor= (int*)     (ws + o); o += 256;
    float*    part  = (float*)   (ws + o); o += (size_t)(NTOK / 256) * 16 * 4;

    prep_kernel<<<64, 256, 0, stream>>>(fc1_w, fc2_w, prompt, w_g,
                                        fc1t, fc2t, plog, pcnt);
    gate_kernel<<<NTOK / 256, 256, 0, stream>>>(x, w_g, plog, xh, gates,
                                                pcnt, part);
    scan_kernel<<<1, 64, 0, stream>>>(pcnt, offs, tileP, cursor);
    place_kernel<<<NTOK / 256, 256, 0, stream>>>(gates, offs, cursor, ent);
    loss_kernel<<<1, 64, 0, stream>>>(part, loss);
    expert_kernel<<<(MAXTILES + 7) / 8, 256, 0, stream>>>(
        xh, gates, ent, pcnt, offs, tileP, fc1t, fc2t, fc1_b, fc2_b, yt);
    transpose_kernel<<<NTOK / 64, 256, 0, stream>>>(yt, y);
}
